// GridNetBlock_76321568850053
// MI455X (gfx1250) — matrix operand order repaired and checked by reading, not compiled, not run on hardware
//
#include <hip/hip_runtime.h>

// ---------------------------------------------------------------------------
// GridNet block for MI455X (gfx1250, wave32, WMMA).
// All GEMMs use v_wmma_f32_16x16x32_f16 (f16 in / f32 accumulate) with LDS
// staging in the exact CDNA5 A/B/C lane layouts (ISA 7.12.2).
// Every GEMM in this model is padded/shaped so M%64==0, K%64==0, N%32==0:
// staging and stores are fully branchless vectorized float4/b64 paths.
// trans variants are compile-time templates (no runtime select in the loop).
// Sequential Mamba scan runs as a register-resident VALU kernel.
// Deterministic (no atomics), graph-capture safe (stream-only ops).
// Needs ~155 MB of d_ws scratch.
// ---------------------------------------------------------------------------

typedef __attribute__((ext_vector_type(16))) _Float16 v16h;
typedef __attribute__((ext_vector_type(8)))  float    v8f;
typedef __attribute__((ext_vector_type(4)))  _Float16 h4;
typedef __attribute__((ext_vector_type(4)))  float    f4;

// ---- problem constants (from the reference) -------------------------------
constexpr int C_DIM  = 64;    // EMB_DIM
constexpr int T_DIM  = 128;   // padded T (== old_T here)
constexpr int Q_DIM  = 128;   // padded Q (== old_Q here)
constexpr int KS     = 4;     // EMB_KS
constexpr int L_SEQ  = 125;   // (Q - KS)/1 + 1
constexpr int S_SEQ  = 128;   // B*T (intra) == B*Q (inter)
constexpr int IN_CH  = 256;   // C_DIM * KS
constexpr int PROJ_N = 552;   // 2*256 + 2*16 + 8 (true width)
constexpr int PROJ_P = 576;   // padded width (multiple of 32)
constexpr int NST    = 16;    // D_STATE
constexpr int NHEAD  = 4;     // attention heads
constexpr int NF     = 128;   // N_FREQS (== Q_DIM)
constexpr int EQ     = 4;     // E_DIM for Q/K
constexpr int EV     = 16;    // CH_V for V
constexpr float EPSF = 1e-5f;

// ---- generic WMMA GEMM ----------------------------------------------------
// C[m][n] = sum_k A'[m][k] * B'[k][n]   (f32 in memory, f16 WMMA compute)
//   TA==0 : A'[m][k] = A[m*lda + k]      TA==1 : A'[m][k] = A[k*lda + m]
//   TB==0 : B'[k][n] = B[k*ldb + n]      TB==1 : B'[k][n] = B[n*ldb + k]
// REQUIRES: M % 64 == 0, K % 64 == 0, N % 32 == 0 (guaranteed by padding).
// batch via blockIdx.z with element strides sA/sB/sC.

template<int TA, int TB>
__global__ __launch_bounds__(256)
void k_gemm(const float* __restrict__ A, const float* __restrict__ B,
            float* __restrict__ C, int K,
            int lda, int ldb, int ldc, long sA, long sB, long sC)
{
  __shared__ _Float16 As[64][72];   // [m][k]
  __shared__ _Float16 Bs[32][72];   // [n][k]

  const int tid   = threadIdx.x;
  const int wave  = tid >> 5;
  const int lane  = tid & 31;
  const int half  = lane >> 4;      // 0: lanes 0-15, 1: lanes 16-31
  const int row16 = lane & 15;      // M (A) / N (B,C) within 16
  const int mi    = wave & 3;       // 4 sub-tiles in M
  const int ni    = wave >> 2;      // 2 sub-tiles in N
  const int m0    = blockIdx.x * 64;
  const int n0    = blockIdx.y * 32;

  A += (long)blockIdx.z * sA;
  B += (long)blockIdx.z * sB;
  C += (long)blockIdx.z * sC;

  v8f acc = {};

  for (int kk = 0; kk < K; kk += 64) {
    // speculative prefetch of next A K-slab (global_prefetch_b8; silently
    // dropped by HW if the address is past the tensor)
    {
      const float* pa = TA ? (A + (long)(kk + 64 + (tid >> 2)) * lda + m0)
                           : (A + (long)(m0 + (tid >> 2)) * lda + kk + 64);
      __builtin_prefetch(pa, 0, 1);
    }

    // ---- stage A tile 64(m) x 64(k): 1024 float4 vectors, 4 / thread ----
#pragma unroll
    for (int i = 0; i < 4; ++i) {
      int v = i * 256 + tid;
      if (TA == 0) {                       // contiguous in k
        int r = v >> 4, cv = v & 15;       // m-row, 16 vectors per row
        f4 d = *(const f4*)(A + (long)(m0 + r) * lda + kk + cv * 4);
        h4 h = { (_Float16)d.x, (_Float16)d.y, (_Float16)d.z, (_Float16)d.w };
        *(h4*)&As[r][cv * 4] = h;
      } else {                             // A[k][m]: contiguous in m
        int kr = v >> 4, cv = v & 15;      // k-row, 16 vectors of m per row
        f4 d = *(const f4*)(A + (long)(kk + kr) * lda + m0 + cv * 4);
        As[cv * 4 + 0][kr] = (_Float16)d.x;
        As[cv * 4 + 1][kr] = (_Float16)d.y;
        As[cv * 4 + 2][kr] = (_Float16)d.z;
        As[cv * 4 + 3][kr] = (_Float16)d.w;
      }
    }
    // ---- stage B tile 32(n) x 64(k): 512 float4 vectors, 2 / thread ----
#pragma unroll
    for (int i = 0; i < 2; ++i) {
      int v = i * 256 + tid;
      if (TB == 0) {                       // B[k][n]: contiguous in n
        int kr = v >> 3, nv = v & 7;       // k-row, 8 vectors of n per row
        f4 d = *(const f4*)(B + (long)(kk + kr) * ldb + n0 + nv * 4);
        Bs[nv * 4 + 0][kr] = (_Float16)d.x;
        Bs[nv * 4 + 1][kr] = (_Float16)d.y;
        Bs[nv * 4 + 2][kr] = (_Float16)d.z;
        Bs[nv * 4 + 3][kr] = (_Float16)d.w;
      } else {                             // B[n][k]: contiguous in k
        int nr = v >> 4, kv = v & 15;      // n-row, 16 vectors of k per row
        f4 d = *(const f4*)(B + (long)(n0 + nr) * ldb + kk + kv * 4);
        h4 h = { (_Float16)d.x, (_Float16)d.y, (_Float16)d.z, (_Float16)d.w };
        *(h4*)&Bs[nr][kv * 4] = h;
      }
    }
    __syncthreads();

    // gather fragments in the CDNA5 WMMA lane layout (ISA 7.12.2)
    const _Float16* arow = &As[mi * 16 + row16][0];
    const _Float16* brow = &Bs[ni * 16 + row16][0];
#pragma unroll
    for (int k0 = 0; k0 < 64; k0 += 32) {
      v16h av, bv;
#pragma unroll
      for (int e = 0; e < 8; ++e) {
        av[e]     = arow[k0 + half * 8 + e];       // VGPR0-3: K = half*8 + e
        av[8 + e] = arow[k0 + 16 + half * 8 + e];  // VGPR4-7: K = 16+half*8+e
      }
#pragma unroll
      for (int e = 0; e < 8; ++e) {
        bv[e]     = brow[k0 + half * 8 + e];       // K = half*8 + e
        bv[8 + e] = brow[k0 + 16 + half * 8 + e];  // K = 16+half*8+e
      }
      acc = __builtin_amdgcn_wmma_f32_16x16x32_f16(
          false, av, false, bv, (short)0, acc, false, false);
    }
    __syncthreads();
  }

  // C/D layout: VGPR r -> M = r + 8*half, N = lane%16 (always in range)
  const int gn = n0 + ni * 16 + row16;
#pragma unroll
  for (int r = 0; r < 8; ++r) {
    int gm = m0 + mi * 16 + r + half * 8;
    C[(long)gm * ldc + gn] = acc[r];
  }
}

// ---- block reduction helper ----------------------------------------------
__device__ __forceinline__ float block_reduce_sum(float v, float* sh) {
  int tid = threadIdx.x;
  sh[tid] = v;
  __syncthreads();
  for (int off = blockDim.x >> 1; off > 0; off >>= 1) {
    if (tid < off) sh[tid] += sh[tid + off];
    __syncthreads();
  }
  float r = sh[0];
  __syncthreads();
  return r;
}

// ---- LayerNorm over channel axis of (C,T,Q) -------------------------------
__global__ void k_ln4d(const float* __restrict__ x, const float* __restrict__ g,
                       const float* __restrict__ b, float* __restrict__ hn)
{
  int idx = blockIdx.x * blockDim.x + threadIdx.x;
  if (idx >= T_DIM * Q_DIM) return;
  int t = idx / Q_DIM, q = idx % Q_DIM;
  float s = 0.f, s2 = 0.f;
  for (int c = 0; c < C_DIM; ++c) {
    float v = x[((long)c * T_DIM + t) * Q_DIM + q];
    s += v; s2 += v * v;
  }
  float mu  = s / C_DIM;
  float var = s2 / C_DIM - mu * mu;
  float inv = rsqrtf(var + EPSF);
  for (int c = 0; c < C_DIM; ++c) {
    long a = ((long)c * T_DIM + t) * Q_DIM + q;
    hn[a] = (x[a] - mu) * inv * g[c] + b[c];
  }
}

// ---- unfold: u[s][l][c*4+k] -----------------------------------------------
// mode 0 (intra): seq along q -> u = hn[c][s][l+k]
// mode 1 (inter): seq along t -> u = hn[c][l+k][s]
__global__ void k_unfold(const float* __restrict__ hn, float* __restrict__ u,
                         int mode)
{
  long idx = (long)blockIdx.x * blockDim.x + threadIdx.x;
  long total = (long)S_SEQ * L_SEQ * IN_CH;
  if (idx >= total) return;
  int ck = idx % IN_CH;
  long sl = idx / IN_CH;
  int l = sl % L_SEQ;
  int s = sl / L_SEQ;
  int c = ck >> 2, k = ck & 3;
  float v;
  if (mode == 0) v = hn[((long)c * T_DIM + s) * Q_DIM + (l + k)];
  else           v = hn[((long)c * T_DIM + (l + k)) * Q_DIM + s];
  u[idx] = v;
}

// ---- RMS norm over last dim (256) with weight -----------------------------
__global__ void k_rms(const float* __restrict__ u, const float* __restrict__ w,
                      float* __restrict__ xn)
{
  __shared__ float sh[256];
  long row = blockIdx.x;               // S*L rows
  int tid = threadIdx.x;
  float v = u[row * IN_CH + tid];
  float ss = block_reduce_sum(v * v, sh);
  float inv = rsqrtf(ss / IN_CH + 1e-5f);
  xn[row * IN_CH + tid] = v * inv * w[tid];
}

// ---- pad in_proj weights [256][552] -> [256][576] (zero-padded) -----------
__global__ void k_pad_inproj(const float* __restrict__ W, float* __restrict__ Bp)
{
  int idx = blockIdx.x * blockDim.x + threadIdx.x;
  if (idx >= IN_CH * PROJ_P) return;
  int row = idx / PROJ_P, col = idx % PROJ_P;
  Bp[idx] = (col < PROJ_N) ? W[row * PROJ_N + col] : 0.f;
}

// ---- Mamba selective scan (register-resident state) -----------------------
// ym[s][l][h*32+p] = (scan_y + D*x) * z,  direction via reverse flag
// proj rows have stride PROJ_P (padded); column offsets are the true ones.
__global__ void k_scan(const float* __restrict__ proj,
                       const float* __restrict__ dt_bias,
                       const float* __restrict__ A_log,
                       const float* __restrict__ D_skip,
                       float* __restrict__ ym, int reverse)
{
  int tid = blockIdx.x * blockDim.x + threadIdx.x;   // 32768 threads
  int p = tid & 31;
  int h = (tid >> 5) & 7;
  int s = tid >> 8;
  float st[NST];
#pragma unroll
  for (int n = 0; n < NST; ++n) st[n] = 0.f;
  const float Aexp = expf(A_log[h]);
  const float db   = dt_bias[h];
  const float Dh   = D_skip[h];
  for (int step = 0; step < L_SEQ; ++step) {
    int l = reverse ? (L_SEQ - 1 - step) : step;
    const float* pr = proj + ((long)s * L_SEQ + l) * PROJ_P;
    float dtr = pr[544 + h] + db;
    float dt  = dtr > 20.f ? dtr : log1pf(expf(dtr));   // softplus
    float a   = expf(-dt * Aexp);
    float xv  = pr[256 + h * 32 + p];
    float dx  = dt * xv;
    float y   = 0.f;
#pragma unroll
    for (int n = 0; n < NST; ++n) {
      st[n] = a * st[n] + dx * pr[512 + n];             // B_t
      y    += st[n] * pr[528 + n];                      // C_t
    }
    y += Dh * xv;
    float z = pr[h * 32 + p];
    ym[((long)s * L_SEQ + l) * IN_CH + h * 32 + p] = y * z;
  }
}

// ---- add residual u into both halves of cat -------------------------------
__global__ void k_addcat(float* __restrict__ cat, const float* __restrict__ u)
{
  long idx = (long)blockIdx.x * blockDim.x + threadIdx.x;
  long total = (long)S_SEQ * L_SEQ * 2 * IN_CH;
  if (idx >= total) return;
  long row = idx / (2 * IN_CH);
  int  j   = idx % (2 * IN_CH);
  cat[idx] += u[row * IN_CH + (j & (IN_CH - 1))];
}

// ---- deconv (overlap-add as deterministic gather) + residual --------------
// mode 0 (intra): xout[c][t][q] = xin + bias[c] + sum_k contrib[(t*L + q-k)][c*4+k]
// mode 1 (inter): xout[c][t][q] = xin + bias[c] + sum_k contrib[(q*L + t-k)][c*4+k]
__global__ void k_deconv(const float* __restrict__ contrib,
                         const float* __restrict__ bias,
                         const float* __restrict__ xin,
                         float* __restrict__ xout, int mode)
{
  long idx = (long)blockIdx.x * blockDim.x + threadIdx.x;
  long total = (long)C_DIM * T_DIM * Q_DIM;
  if (idx >= total) return;
  int q = idx % Q_DIM;
  int t = (idx / Q_DIM) % T_DIM;
  int c = idx / (Q_DIM * T_DIM);
  int s   = (mode == 0) ? t : q;
  int pos = (mode == 0) ? q : t;
  float acc = bias[c];
#pragma unroll
  for (int k = 0; k < KS; ++k) {
    int l = pos - k;
    if (l >= 0 && l < L_SEQ)
      acc += contrib[((long)s * L_SEQ + l) * IN_CH + c * 4 + k];
  }
  xout[idx] = xin[idx] + acc;
}

// ---- head projection: leaky-relu + groupnorm over (e,f) per (h,t) ---------
// yt is [(t*128+f)][ldy]; writes Of[h][t][e*128+f]
__global__ void k_head_norm(const float* __restrict__ yt,
                            const float* __restrict__ bias,
                            const float* __restrict__ alpha,
                            const float* __restrict__ g,
                            const float* __restrict__ bt,
                            float* __restrict__ Of, int E, int ldy)
{
  __shared__ float sh[256];
  int h = blockIdx.x / T_DIM;
  int t = blockIdx.x % T_DIM;
  int EF = E * NF;
  float al = alpha[h];
  float s = 0.f, s2 = 0.f;
  for (int i = threadIdx.x; i < EF; i += blockDim.x) {
    int e = i / NF, f = i % NF;
    float v = yt[((long)t * NF + f) * ldy + h * E + e] + bias[h * E + e];
    float lr = v >= 0.f ? v : al * v;
    s += lr; s2 += lr * lr;
  }
  float sum  = block_reduce_sum(s, sh);
  float sum2 = block_reduce_sum(s2, sh);
  float mu  = sum / EF;
  float var = sum2 / EF - mu * mu;
  float inv = rsqrtf(var + EPSF);
  for (int i = threadIdx.x; i < EF; i += blockDim.x) {
    int e = i / NF, f = i % NF;
    float v = yt[((long)t * NF + f) * ldy + h * E + e] + bias[h * E + e];
    float lr = v >= 0.f ? v : al * v;
    Of[((long)h * T_DIM + t) * EF + e * NF + f] =
        (lr - mu) * inv * g[(h * E + e) * NF + f] + bt[(h * E + e) * NF + f];
  }
}

// ---- softmax over last dim of S1[h][t][128], scale 1/sqrt(512) ------------
__global__ void k_softmax(float* __restrict__ S1)
{
  __shared__ float sh[128];
  long row = blockIdx.x;                 // NHEAD*T rows
  int tid = threadIdx.x;                 // 128 threads
  float* r = S1 + row * T_DIM;
  const float scale = rsqrtf(512.f);
  float v = r[tid] * scale;
  sh[tid] = v; __syncthreads();
  for (int off = 64; off > 0; off >>= 1) {
    if (tid < off) sh[tid] = fmaxf(sh[tid], sh[tid + off]);
    __syncthreads();
  }
  float mx = sh[0]; __syncthreads();
  float e = expf(v - mx);
  float ssum = block_reduce_sum(e, sh);
  r[tid] = e / ssum;
}

// ---- repack W(h,e,c) -> B[c][n] zero-padded to NP columns -----------------
__global__ void k_repack_whec(const float* __restrict__ W, float* __restrict__ Bm,
                              int HE, int NP)
{
  int idx = blockIdx.x * blockDim.x + threadIdx.x;
  if (idx >= C_DIM * NP) return;
  int c = idx / NP, n = idx % NP;
  Bm[idx] = (n < HE) ? W[n * C_DIM + c] : 0.f;
}

// ---- repack attention output oH[h][t][cv*128+f] -> av[(h*16+cv)][t*128+f] -
__global__ void k_repack_attn(const float* __restrict__ oH, float* __restrict__ av)
{
  long idx = (long)blockIdx.x * blockDim.x + threadIdx.x;
  long total = (long)NHEAD * T_DIM * EV * NF;
  if (idx >= total) return;
  int h = idx / ((long)T_DIM * EV * NF);
  long r = idx % ((long)T_DIM * EV * NF);
  int t = r / (EV * NF);
  int j = r % (EV * NF);
  int cv = j / NF, f = j % NF;
  av[((long)(h * EV + cv) * T_DIM + t) * NF + f] = oH[idx];
}

// ---- final: leaky + norm over (c,f) per t, + gamma/beta + residual --------
__global__ void k_final_norm(const float* __restrict__ yt,
                             const float* __restrict__ bp,
                             const float* __restrict__ ap,
                             const float* __restrict__ gp,
                             const float* __restrict__ btp,
                             const float* __restrict__ res,
                             float* __restrict__ out)
{
  __shared__ float sh[256];
  int t = blockIdx.x;
  int CF = C_DIM * NF;                   // 8192
  float al = ap[0];
  float s = 0.f, s2 = 0.f;
  for (int i = threadIdx.x; i < CF; i += blockDim.x) {
    int o = i / NF, f = i % NF;
    float v = yt[((long)t * NF + f) * C_DIM + o] + bp[o];
    float lr = v >= 0.f ? v : al * v;
    s += lr; s2 += lr * lr;
  }
  float sum  = block_reduce_sum(s, sh);
  float sum2 = block_reduce_sum(s2, sh);
  float mu  = sum / CF;
  float var = sum2 / CF - mu * mu;
  float inv = rsqrtf(var + EPSF);
  for (int i = threadIdx.x; i < CF; i += blockDim.x) {
    int o = i / NF, f = i % NF;
    float v = yt[((long)t * NF + f) * C_DIM + o] + bp[o];
    float lr = v >= 0.f ? v : al * v;
    long a = ((long)o * T_DIM + t) * NF + f;
    out[a] = (lr - mu) * inv * gp[o * NF + f] + btp[o * NF + f] + res[a];
  }
}

// ---------------------------------------------------------------------------
extern "C" void kernel_launch(void* const* d_in, const int* in_sizes, int n_in,
                              void* d_out, int out_size, void* d_ws, size_t ws_size,
                              hipStream_t stream)
{
  (void)in_sizes; (void)n_in; (void)out_size; (void)ws_size;
  auto F = [&](int i) { return (const float*)d_in[i]; };

  // input order: x, intra_norm_g/b, inter_norm_g/b,
  // m_intra_f{norm_w,in_proj,dt_bias,A_log,D_skip,out_proj}(5..10), m_intra_b(11..16),
  // m_inter_f(17..22), m_inter_b(23..28), intra_lin_W(29), intra_lin_b(30),
  // inter_lin_W(31), inter_lin_b(32), WQ..btQ(33..37), WK..btK(38..42),
  // WV..btV(43..47), Wp(48), bp(49), ap(50), gp(51), btp(52)
  const float* x = F(0);

  // ---- workspace partition (floats), sequential reuse ----
  float* W = (float*)d_ws;
  size_t off = 0;
  auto alloc = [&](size_t n) { float* p = W + off; off += (n + 63) & ~(size_t)63; return p; };
  float* hn   = alloc((size_t)C_DIM * T_DIM * Q_DIM);      // 1.05M
  float* u    = alloc((size_t)S_SEQ * L_SEQ * IN_CH);      // 4.10M
  float* xn   = alloc((size_t)S_SEQ * L_SEQ * IN_CH);      // 4.10M
  float* proj = alloc((size_t)S_SEQ * L_SEQ * PROJ_P);     // 9.22M (reused as contrib)
  float* ym   = alloc((size_t)S_SEQ * L_SEQ * IN_CH);      // 4.10M
  float* cat  = alloc((size_t)S_SEQ * L_SEQ * 2 * IN_CH);  // 8.19M
  float* x1   = alloc((size_t)C_DIM * T_DIM * Q_DIM);
  float* x2   = alloc((size_t)C_DIM * T_DIM * Q_DIM);
  float* Qf   = alloc((size_t)NHEAD * T_DIM * EQ * NF);    // 262k
  float* Kf   = alloc((size_t)NHEAD * T_DIM * EQ * NF);
  float* Vf   = alloc((size_t)NHEAD * T_DIM * EV * NF);    // 1.05M
  float* yt   = alloc((size_t)T_DIM * NF * C_DIM);         // 1.05M (head gemm out, reused)
  float* S1m  = alloc((size_t)NHEAD * T_DIM * T_DIM);      // 65k
  float* oH   = alloc((size_t)NHEAD * T_DIM * EV * NF);    // 1.05M
  float* av   = alloc((size_t)C_DIM * T_DIM * NF);         // 1.05M
  float* Bp   = alloc((size_t)IN_CH * PROJ_P);             // padded in_proj weights
  float* Bq   = alloc((size_t)C_DIM * 32);                 // padded (N=16 -> 32)
  float* Bk   = alloc((size_t)C_DIM * 32);
  float* Bv   = alloc((size_t)C_DIM * 64);
  float* contrib = proj;                                   // alias (proj dead by then)

  auto gemm = [&](const float* A, const float* B, float* C, int M, int N, int K,
                  int lda, int ldb, int ldc, int tA, int tB,
                  long sA, long sB, long sC, int batch) {
    dim3 grid(M / 64, N / 32, batch);   // N % 32 == 0 by construction
    dim3 blk(256);
    if (!tA && !tB)
      k_gemm<0,0><<<grid, blk, 0, stream>>>(A, B, C, K, lda, ldb, ldc, sA, sB, sC);
    else if (tA && !tB)
      k_gemm<1,0><<<grid, blk, 0, stream>>>(A, B, C, K, lda, ldb, ldc, sA, sB, sC);
    else if (!tA && tB)
      k_gemm<0,1><<<grid, blk, 0, stream>>>(A, B, C, K, lda, ldb, ldc, sA, sB, sC);
    else
      k_gemm<1,1><<<grid, blk, 0, stream>>>(A, B, C, K, lda, ldb, ldc, sA, sB, sC);
  };

  const int SL = S_SEQ * L_SEQ;   // 16000
  auto cdiv = [](long n, int b) { return (int)((n + b - 1) / b); };

  // ---- one intra/inter stage ----
  auto stage = [&](const float* xin, const float* lng, const float* lnb,
                   int pf, int pb, const float* linW, const float* linb,
                   float* xout, int mode) {
    k_ln4d<<<cdiv(T_DIM * Q_DIM, 256), 256, 0, stream>>>(xin, lng, lnb, hn);
    k_unfold<<<cdiv((long)SL * IN_CH, 256), 256, 0, stream>>>(hn, u, mode);
    for (int d = 0; d < 2; ++d) {
      int base = d ? pb : pf;
      k_rms<<<SL, 256, 0, stream>>>(u, F(base + 0), xn);
      k_pad_inproj<<<cdiv(IN_CH * PROJ_P, 256), 256, 0, stream>>>(F(base + 1), Bp);
      gemm(xn, Bp, proj, SL, PROJ_P, IN_CH, IN_CH, PROJ_P, PROJ_P,
           0, 0, 0, 0, 0, 1);                                  // in_proj (padded)
      k_scan<<<128, 256, 0, stream>>>(proj, F(base + 2), F(base + 3),
                                      F(base + 4), ym, d);
      gemm(ym, F(base + 5), cat + (d ? IN_CH : 0), SL, IN_CH, IN_CH,
           IN_CH, IN_CH, 2 * IN_CH, 0, 0, 0, 0, 0, 1);          // out_proj
    }
    k_addcat<<<cdiv((long)SL * 2 * IN_CH, 256), 256, 0, stream>>>(cat, u);
    gemm(cat, linW, contrib, SL, IN_CH, 2 * IN_CH,
         2 * IN_CH, IN_CH, IN_CH, 0, 0, 0, 0, 0, 1);            // deconv GEMM
    k_deconv<<<cdiv((long)C_DIM * T_DIM * Q_DIM, 256), 256, 0, stream>>>(
        contrib, linb, xin, xout, mode);
  };

  stage(x,  F(1), F(2), 5, 11, F(29), F(30), x1, 0);   // intra (seq along Q)
  stage(x1, F(3), F(4), 17, 23, F(31), F(32), x2, 1);  // inter (seq along T)

  // ---- attention heads ----
  const int MTF = T_DIM * NF;   // 16384
  k_repack_whec<<<cdiv(C_DIM * 32, 256), 256, 0, stream>>>(F(33), Bq, NHEAD * EQ, 32);
  k_repack_whec<<<cdiv(C_DIM * 32, 256), 256, 0, stream>>>(F(38), Bk, NHEAD * EQ, 32);
  k_repack_whec<<<cdiv(C_DIM * 64, 256), 256, 0, stream>>>(F(43), Bv, NHEAD * EV, 64);

  // Q/K/V: x2 is [c][(t,f)] -> transA GEMM, then leaky+norm into [h][t][e*F+f]
  gemm(x2, Bq, yt, MTF, 32, C_DIM, MTF, 32, 32, 1, 0, 0, 0, 0, 1);
  k_head_norm<<<NHEAD * T_DIM, 256, 0, stream>>>(yt, F(34), F(35), F(36), F(37), Qf, EQ, 32);
  gemm(x2, Bk, yt, MTF, 32, C_DIM, MTF, 32, 32, 1, 0, 0, 0, 0, 1);
  k_head_norm<<<NHEAD * T_DIM, 256, 0, stream>>>(yt, F(39), F(40), F(41), F(42), Kf, EQ, 32);
  gemm(x2, Bv, yt, MTF, 64, C_DIM, MTF, 64, 64, 1, 0, 0, 0, 0, 1);
  k_head_norm<<<NHEAD * T_DIM, 256, 0, stream>>>(yt, F(44), F(45), F(46), F(47), Vf, EV, 64);

  // attn = softmax(Q K^T / sqrt(512)) ; out = attn V   (batched over heads)
  const int DQK = EQ * NF;   // 512
  const int DV  = EV * NF;   // 2048
  gemm(Qf, Kf, S1m, T_DIM, T_DIM, DQK, DQK, DQK, T_DIM, 0, 1,
       (long)T_DIM * DQK, (long)T_DIM * DQK, (long)T_DIM * T_DIM, NHEAD);
  k_softmax<<<NHEAD * T_DIM, 128, 0, stream>>>(S1m);
  gemm(S1m, Vf, oH, T_DIM, DV, T_DIM, T_DIM, DV, DV, 0, 0,
       (long)T_DIM * T_DIM, (long)T_DIM * DV, (long)T_DIM * DV, NHEAD);

  // final projection + norm + residual
  k_repack_attn<<<cdiv((long)NHEAD * T_DIM * DV, 256), 256, 0, stream>>>(oH, av);
  gemm(av, F(48), yt, MTF, C_DIM, C_DIM, MTF, C_DIM, C_DIM, 1, 1, 0, 0, 0, 1);
  k_final_norm<<<T_DIM, 256, 0, stream>>>(yt, F(49), F(50), F(51), F(52),
                                          x2, (float*)d_out);
}
